// InteractionDecoder_85796266705058
// MI455X (gfx1250) — matrix operand order repaired and checked by reading, not compiled, not run on hardware
//
#include <hip/hip_runtime.h>
#include <hip/hip_bf16.h>

typedef __attribute__((ext_vector_type(16))) _Float16 v16h;
typedef __attribute__((ext_vector_type(8)))  _Float16 v8h;
typedef __attribute__((ext_vector_type(4)))  _Float16 v4h;
typedef __attribute__((ext_vector_type(8)))  float    v8f;
typedef __attribute__((ext_vector_type(4)))  unsigned v4u;
typedef __attribute__((ext_vector_type(8)))  int      v8i;
typedef __attribute__((ext_vector_type(4)))  int      v4i;

#define KD   128
#define KA   300
#define KK   6
#define KH   8
#define KT   50
#define KPL  1500
#define KFUT 60
#define KREC 3
#define KL   2
#define KM   (KA*KK)     /* 1800  */
#define KNT  (KA*KT)     /* 15000 */

#if defined(__has_builtin)
#if __has_builtin(__builtin_amdgcn_tensor_load_to_lds)
#define USE_TDM 1
#endif
#endif
#ifndef USE_TDM
#define USE_TDM 0
#endif

__device__ __forceinline__ unsigned fenc(float f){
  unsigned u = __float_as_uint(f);
  return (u & 0x80000000u) ? ~u : (u | 0x80000000u);
}
__device__ __forceinline__ float fdec(unsigned u){
  return __uint_as_float((u & 0x80000000u) ? (u & 0x7fffffffu) : ~u);
}

#if USE_TDM
// Build a 2D Tensor-DMA descriptor (D#) and issue TENSOR_LOAD_TO_LDS.
// D# group0: [1:0]=count=1, [63:32]=lds_addr, [120:64]=global_addr, [127:126]=type(2)
// D# group1: [17:16]=data_size(2 => 4B), [79:48]=tensor_dim0, [111:80]=tensor_dim1,
//            [127:112]=tile_dim0, [143:128]=tile_dim1, [207:160]=tensor_dim0_stride
// OOB elements (beyond tensor_dim0/1) are zero-filled -> edge tiles need no guards.
__device__ __forceinline__ void tdm_load_2d(unsigned lds_off, const float* gaddr,
                                            unsigned width_rem, unsigned height_rem,
                                            unsigned stride_elems,
                                            unsigned tile_w, unsigned tile_h)
{
  unsigned long long ga = (unsigned long long)(size_t)gaddr;
  v4u g0;
  g0[0] = 1u;                                              // count=1 (valid user D#)
  g0[1] = lds_off;                                         // LDS byte address
  g0[2] = (unsigned)(ga & 0xffffffffu);                    // global_addr[31:0]
  g0[3] = (unsigned)((ga >> 32) & 0x01ffffffu) | (2u << 30); // addr[56:32] | type=2
  v8i g1;
  g1[0] = (int)(2u << 16);                                 // workgroup_mask=0, data_size=4B
  g1[1] = (int)((width_rem & 0xffffu) << 16);              // tensor_dim0[15:0]
  g1[2] = (int)(((width_rem >> 16) & 0xffffu) | ((height_rem & 0xffffu) << 16));
  g1[3] = (int)(((height_rem >> 16) & 0xffffu) | (tile_w << 16)); // tile_dim0
  g1[4] = (int)tile_h;                                     // tile_dim1 (tile_dim2=0)
  g1[5] = (int)stride_elems;                               // tensor_dim0_stride[31:0]
  g1[6] = 0;                                               // stride hi, dim1_stride=0
  g1[7] = 0;
  v4i z4 = {0, 0, 0, 0};
#if __clang_major__ >= 23
  v8i z8 = {0, 0, 0, 0, 0, 0, 0, 0};
  __builtin_amdgcn_tensor_load_to_lds(g0, g1, z4, z4, z8, 0);
#else
  __builtin_amdgcn_tensor_load_to_lds(g0, g1, z4, z4, 0);
#endif
}
#endif

// ===================== WMMA GEMM =====================
// C[M,N] = act(A[M,K] * B[K,N] + bias)   (fp32 in/out, f16 WMMA, f32 accum)
// ACT: 0 none, 1 relu, 2 sigmoid.  ACC: C += result.
// Block tile: 16(M) x 64(N), 4 waves, K stepped by 32.
#if USE_TDM
// TDM variant: wave0 DMAs the A (16x32) and B (32x64) f32 tiles into LDS via the
// Tensor Data Mover (zero-fill OOB -> no edge guards), waits TENSORcnt==0, then
// the workgroup barrier publishes LDS; fragments convert f32->f16 at read time.
template<int ACT, bool ACC>
__global__ __launch_bounds__(128) void gemm_wmma_k(
    const float* __restrict__ Ap, int lda,
    const float* __restrict__ Bp, int ldb,
    const float* __restrict__ bias,
    float* __restrict__ Cp, int ldc,
    int M, int N, int K)
{
  __shared__ float Asf[16][32];
  __shared__ float Bsf[32][64];
  const int tid  = threadIdx.x;
  const int lane = tid & 31;
  const int wave = tid >> 5;
  const int m0 = blockIdx.y * 16;
  const int n0 = blockIdx.x * 64;
  v8f acc = {};
  const int arow  = lane & 15;
  const int koffA = (lane >= 16) ? 8 : 0;    // 16-bit A 16x32 fragment layout
  const int bcol  = (lane & 15) + wave * 16; // 16-bit B 32x16 fragment layout
  const int koffB = (lane >= 16) ? 16 : 0;
  const unsigned ldsA = (unsigned)(size_t)(void*)&Asf[0][0];
  const unsigned ldsB = (unsigned)(size_t)(void*)&Bsf[0][0];
  // scalar (per-wave uniform) issuer flag so non-issuing waves s_cbranch around TDM
  const int issuer = (__builtin_amdgcn_readfirstlane(wave) == 0);
  for (int k0 = 0; k0 < K; k0 += 32) {
    if (issuer) {
      // A tile: x=k (width 32), y=m (height 16), row stride lda
      tdm_load_2d(ldsA, Ap + (size_t)m0 * lda + k0,
                  (unsigned)(K - k0), (unsigned)(M - m0), (unsigned)lda, 32u, 16u);
      // B tile: x=n (width 64), y=k (height 32), row stride ldb
      tdm_load_2d(ldsB, Bp + (size_t)k0 * ldb + n0,
                  (unsigned)(N - n0), (unsigned)(K - k0), (unsigned)ldb, 64u, 32u);
      __builtin_amdgcn_s_wait_tensorcnt(0);
    }
    __syncthreads();
    v16h av, bv;
#pragma unroll
    for (int i = 0; i < 16; i++) {
      int ka = (i & 7) + ((i >= 8) ? 16 : 0) + koffA;
      av[i] = (_Float16)Asf[arow][ka];
      bv[i] = (_Float16)Bsf[ka][bcol];
    }
    acc = __builtin_amdgcn_wmma_f32_16x16x32_f16(false, av, false, bv,
                                                 (short)0, acc, false, false);
    __syncthreads();   // protect LDS against next k-step's DMA
  }
  const int n  = n0 + wave * 16 + (lane & 15);
  const int mb = m0 + ((lane >= 16) ? 8 : 0);
  if (n < N) {
    const float bn = bias ? bias[n] : 0.f;
#pragma unroll
    for (int j = 0; j < 8; j++) {
      int m = mb + j;
      if (m < M) {
        float v = acc[j] + bn;
        if (ACT == 1) v = v > 0.f ? v : 0.f;
        if (ACT == 2) v = 1.f / (1.f + __expf(-v));
        size_t o = (size_t)m * ldc + n;
        if (ACC) Cp[o] += v; else Cp[o] = v;
      }
    }
  }
}
#else
// Sync fallback: packed f32->f16 staging, ds_store_b64,
// ds_load_b128 fragment reads; guarded loops only on edge tiles.
template<int ACT, bool ACC>
__global__ __launch_bounds__(128) void gemm_wmma_k(
    const float* __restrict__ Ap, int lda,
    const float* __restrict__ Bp, int ldb,
    const float* __restrict__ bias,
    float* __restrict__ Cp, int ldc,
    int M, int N, int K)
{
  __shared__ _Float16 As[16][40];
  __shared__ _Float16 Bs[64][40];
  const int tid  = threadIdx.x;
  const int lane = tid & 31;
  const int wave = tid >> 5;
  const int m0 = blockIdx.y * 16;
  const int n0 = blockIdx.x * 64;
  v8f acc = {};
  const int arow  = lane & 15;
  const int koffA = (lane >= 16) ? 8 : 0;
  const int bcol  = (lane & 15) + wave * 16;
  const int koffB = (lane >= 16) ? 8 : 0;
  const bool fullM = (m0 + 16 <= M);
  const bool fullN = (n0 + 64 <= N);
  for (int k0 = 0; k0 < K; k0 += 32) {
    const bool fullK = (k0 + 32 <= K);
    if (fullM & fullK) {
      const int r = tid >> 3, c = (tid & 7) * 4;
      const float* ap = Ap + (size_t)(m0 + r) * lda + (k0 + c);
      v4h h;
      h[0] = (_Float16)ap[0]; h[1] = (_Float16)ap[1];
      h[2] = (_Float16)ap[2]; h[3] = (_Float16)ap[3];
      *(v4h*)&As[r][c] = h;
      if (k0 + 32 < K) __builtin_prefetch(ap + 32, 0, 1);
    } else {
      for (int t = tid; t < 16 * 32; t += 128) {
        int r = t >> 5, c = t & 31;
        int gm = m0 + r, gk = k0 + c;
        As[r][c] = (_Float16)((gm < M && gk < K) ? Ap[(size_t)gm * lda + gk] : 0.f);
      }
    }
    if (fullN & fullK) {
#pragma unroll
      for (int j = 0; j < 4; j++) {
        const int task = tid + j * 128;
        const int n = task & 63, kg = task >> 6;
        const float* bp = Bp + (size_t)(k0 + kg * 4) * ldb + (n0 + n);
        v4h h;
        h[0] = (_Float16)bp[0];
        h[1] = (_Float16)bp[(size_t)ldb];
        h[2] = (_Float16)bp[2 * (size_t)ldb];
        h[3] = (_Float16)bp[3 * (size_t)ldb];
        *(v4h*)&Bs[n][kg * 4] = h;
        if (j == 0 && k0 + 32 < K) __builtin_prefetch(bp + 32 * (size_t)ldb, 0, 1);
      }
    } else {
      for (int t = tid; t < 32 * 64; t += 128) {
        int r = t >> 6, c = t & 63;
        int gk = k0 + r, gn = n0 + c;
        Bs[c][r] = (_Float16)((gk < K && gn < N) ? Bp[(size_t)gk * ldb + gn] : 0.f);
      }
    }
    __syncthreads();
    v8h a0 = *(const v8h*)&As[arow][koffA];
    v8h a1 = *(const v8h*)&As[arow][koffA + 16];
    v8h b0 = *(const v8h*)&Bs[bcol][koffB];
    v8h b1 = *(const v8h*)&Bs[bcol][koffB + 16];
    v16h av, bv;
#pragma unroll
    for (int i = 0; i < 8; i++) {
      av[i] = a0[i]; av[i + 8] = a1[i];
      bv[i] = b0[i]; bv[i + 8] = b1[i];
    }
    acc = __builtin_amdgcn_wmma_f32_16x16x32_f16(false, av, false, bv,
                                                 (short)0, acc, false, false);
    __syncthreads();
  }
  const int n  = n0 + wave * 16 + (lane & 15);
  const int mb = m0 + ((lane >= 16) ? 8 : 0);
  if (n < N) {
    const float bn = bias ? bias[n] : 0.f;
#pragma unroll
    for (int j = 0; j < 8; j++) {
      int m = mb + j;
      if (m < M) {
        float v = acc[j] + bn;
        if (ACT == 1) v = v > 0.f ? v : 0.f;
        if (ACT == 2) v = 1.f / (1.f + __expf(-v));
        size_t o = (size_t)m * ldc + n;
        if (ACC) Cp[o] += v; else Cp[o] = v;
      }
    }
  }
}
#endif

// ===================== LayerNorm over last dim 128 =====================
__global__ __launch_bounds__(128) void ln128_k(
    const float* __restrict__ x, const float* __restrict__ w,
    const float* __restrict__ b, float* __restrict__ y, int relu)
{
  __shared__ float red[128];
  const int row = blockIdx.x, d = threadIdx.x;
  float v = x[(size_t)row * 128 + d];
  red[d] = v; __syncthreads();
  for (int s = 64; s > 0; s >>= 1) { if (d < s) red[d] += red[d + s]; __syncthreads(); }
  float mu = red[0] * (1.f / 128.f);
  __syncthreads();
  float c = v - mu;
  red[d] = c * c; __syncthreads();
  for (int s = 64; s > 0; s >>= 1) { if (d < s) red[d] += red[d + s]; __syncthreads(); }
  float var = red[0] * (1.f / 128.f);
  float o = c * rsqrtf(var + 1e-5f) * w[d] + b[d];
  if (relu) o = o > 0.f ? o : 0.f;
  y[(size_t)row * 128 + d] = o;
}

// ===================== elementwise helpers =====================
__global__ void fill_k(float* p, long n){
  long i = blockIdx.x * (long)blockDim.x + threadIdx.x;
  if (i < n) p[i] = 0.f;
}
__global__ void tile_rows_k(const float* __restrict__ src, float* __restrict__ dst,
                            int period, int stride, int off, long n){
  long i = blockIdx.x * (long)blockDim.x + threadIdx.x;
  if (i >= n) return;
  int row = (int)(i >> 7), d = (int)(i & 127);
  dst[i] = src[(size_t)(row % period) * stride + off + d];
}
__global__ void add_k(const float* __restrict__ a, const float* __restrict__ b,
                      float* __restrict__ o, long n){
  long i = blockIdx.x * (long)blockDim.x + threadIdx.x;
  if (i < n) o[i] = a[i] + b[i];
}
__global__ void elu11_k(const float* __restrict__ a, float* __restrict__ o, long n){
  long i = blockIdx.x * (long)blockDim.x + threadIdx.x;
  if (i >= n) return;
  float x = a[i];
  o[i] = (x > 0.f ? x : __expf(x) - 1.f) + 1.1f;
}
__global__ void cat2_k(const float* __restrict__ a, const float* __restrict__ s,
                       float* __restrict__ cat, long n){
  long i = blockIdx.x * (long)blockDim.x + threadIdx.x;
  if (i >= n) return;
  int m = (int)(i >> 8), c = (int)(i & 255);
  cat[i] = (c < 128) ? a[(size_t)m * 128 + c] : s[(size_t)m * 128 + (c - 128)];
}
__global__ void gate_o_k(const float* __restrict__ agg, const float* __restrict__ g,
                         const float* __restrict__ s, float* __restrict__ o, long n){
  long i = blockIdx.x * (long)blockDim.x + threadIdx.x;
  if (i < n) o[i] = agg[i] + g[i] * (s[i] - agg[i]);
}

// ===================== Fourier feature generation =====================
__global__ void four_feat_k(const float* __restrict__ X, int din, int dim,
                            const float* __restrict__ freqs,
                            float* __restrict__ feat, int base, long n){
  long i = blockIdx.x * (long)blockDim.x + threadIdx.x;
  if (i >= n) return;
  int r = (int)(i / 129), c = (int)(i % 129);
  float x = X[(size_t)(base + r) * din + dim];
  float v;
  if (c < 64)       v = __cosf(6.2831853071795864f * x * freqs[dim * 64 + c]);
  else if (c < 128) v = __sinf(6.2831853071795864f * x * freqs[dim * 64 + (c - 64)]);
  else              v = x;
  feat[i] = v;
}

// ===================== edge attention pieces =====================
__global__ void edge_sim_k(const float* __restrict__ q, const float* __restrict__ k,
                           const float* __restrict__ rk,
                           const int* __restrict__ src, const int* __restrict__ dst,
                           float* __restrict__ sim, unsigned* __restrict__ umax, long EH){
  long i = blockIdx.x * (long)blockDim.x + threadIdx.x;
  if (i >= EH) return;
  int e = (int)(i >> 3), h = (int)(i & 7);
  int s = src[e], d = dst[e];
  const float* qp = q  + (size_t)d * 128 + h * 16;
  const float* kp = k  + (size_t)s * 128 + h * 16;
  const float* rp = rk + (size_t)e * 128 + h * 16;
  float a = 0.f;
#pragma unroll
  for (int j = 0; j < 16; j++) a += qp[j] * (kp[j] + rp[j]);
  a *= 0.25f;  // 1/sqrt(16)
  sim[i] = a;
  atomicMax(umax + (size_t)d * 8 + h, fenc(a));
}
__global__ void edge_exp_k(const float* __restrict__ sim, const unsigned* __restrict__ umax,
                           const int* __restrict__ dst,
                           float* __restrict__ ex, float* __restrict__ den, long EH){
  long i = blockIdx.x * (long)blockDim.x + threadIdx.x;
  if (i >= EH) return;
  int e = (int)(i >> 3), h = (int)(i & 7);
  int d = dst[e];
  float m = fdec(umax[(size_t)d * 8 + h]);
  float v = __expf(sim[i] - m);
  ex[i] = v;
  atomicAdd(den + (size_t)d * 8 + h, v);
}
__global__ void edge_scatter_k(const float* __restrict__ ex, const float* __restrict__ den,
                               const float* __restrict__ v, const float* __restrict__ rv,
                               const int* __restrict__ src, const int* __restrict__ dst,
                               float* __restrict__ agg, long EH){
  long i = blockIdx.x * (long)blockDim.x + threadIdx.x;
  if (i >= EH) return;
  int e = (int)(i >> 3), h = (int)(i & 7);
  int s = src[e], d = dst[e];
  float alpha = ex[i] / den[(size_t)d * 8 + h];
  const float* vp = v  + (size_t)s * 128 + h * 16;
  const float* rp = rv + (size_t)e * 128 + h * 16;
  float* ap = agg + (size_t)d * 128 + h * 16;
#pragma unroll
  for (int j = 0; j < 16; j++) atomicAdd(ap + j, alpha * (vp[j] + rp[j]));
}

// ===================== dense mode-to-mode attention =====================
__global__ void m2m_attn_k(const float* __restrict__ q, const float* __restrict__ k,
                           const float* __restrict__ v, float* __restrict__ agg){
  int i = blockIdx.x * blockDim.x + threadIdx.x;
  if (i >= KA * KK * KH) return;
  int a  = i / (KK * KH);
  int r  = i % (KK * KH);
  int kq = r / KH, h = r % KH;
  const float* qp = q + ((size_t)(a * KK + kq) * 128 + h * 16);
  float sim[KK], mx = -1e30f;
#pragma unroll
  for (int l = 0; l < KK; l++) {
    const float* kp = k + ((size_t)(a * KK + l) * 128 + h * 16);
    float s = 0.f;
#pragma unroll
    for (int j = 0; j < 16; j++) s += qp[j] * kp[j];
    s *= 0.25f;
    sim[l] = s;
    mx = fmaxf(mx, s);
  }
  float dn = 0.f;
#pragma unroll
  for (int l = 0; l < KK; l++) { sim[l] = __expf(sim[l] - mx); dn += sim[l]; }
  float out[16];
#pragma unroll
  for (int j = 0; j < 16; j++) out[j] = 0.f;
#pragma unroll
  for (int l = 0; l < KK; l++) {
    float w = sim[l] / dn;
    const float* vp = v + ((size_t)(a * KK + l) * 128 + h * 16);
#pragma unroll
    for (int j = 0; j < 16; j++) out[j] += w * vp[j];
  }
  float* ap = agg + ((size_t)(a * KK + kq) * 128 + h * 16);
#pragma unroll
  for (int j = 0; j < 16; j++) ap[j] = out[j];
}

// ===================== GRU pointwise gate =====================
__global__ void gru_gate_k(const float* __restrict__ gi, const float* __restrict__ gh,
                           float* __restrict__ h, long n){
  long i = blockIdx.x * (long)blockDim.x + threadIdx.x;
  if (i >= n) return;
  int m = (int)(i >> 7), d = (int)(i & 127);
  size_t b = (size_t)m * 384;
  float ir = gi[b + d],       hr = gh[b + d];
  float iz = gi[b + 128 + d], hz = gh[b + 128 + d];
  float in = gi[b + 256 + d], hn = gh[b + 256 + d];
  float rr = 1.f / (1.f + __expf(-(ir + hr)));
  float zz = 1.f / (1.f + __expf(-(iz + hz)));
  float nn = tanhf(in + rr * hn);
  h[i] = (1.f - zz) * nn + zz * h[i];
}

// ===================== cumsum over FUT =====================
__global__ void cumsum_k(const float* __restrict__ in, float* __restrict__ out,
                         int mode, int rows){
  int i = blockIdx.x * blockDim.x + threadIdx.x;
  if (i >= rows * 2) return;
  int m = i >> 1, c = i & 1;
  float s = 0.f;
  for (int t = 0; t < KFUT; t++) {
    float x = in[(size_t)m * (KFUT * 2) + 2 * t + c];
    if (mode) x = (x > 0.f ? x : __expf(x) - 1.f) + 1.f;
    s += x;
    out[(size_t)m * (KFUT * 2) + 2 * t + c] = mode ? s + 0.1f : s;
  }
}

// ===================== host-side parameter structs =====================
struct Lin  { const float *w, *b; };
struct LnP  { const float *w, *b; };
struct AttnP {
  Lin f1, f2, g; const float *k, *kr;
  LnP ln_d, ln_f, ln_r, ln_s;
  Lin o, q, s, v, vr;
};
struct FourP {
  const float *b1, *b2, *freqs, *lnb; LnP lno; const float *lnw;
  Lin out; const float *w1, *w2; int din;
};
struct MlpP { Lin l1, l2; LnP ln; };
struct GruP { const float *bh, *bi, *h0, *wh, *wi; };

struct Cur {
  void* const* din; int i;
  const float* nf(){ return (const float*)din[i++]; }
};
static void fillLin(Cur& c, Lin& l){ l.b = c.nf(); l.w = c.nf(); }
static void fillLn (Cur& c, LnP& l){ l.w = c.nf(); l.b = c.nf(); }
static void fillAttn(Cur& c, AttnP& a, bool has_r){
  fillLin(c, a.f1); fillLin(c, a.f2); fillLin(c, a.g);
  a.k = c.nf();
  if (has_r) a.kr = c.nf();
  fillLn(c, a.ln_d); fillLn(c, a.ln_f);
  if (has_r) fillLn(c, a.ln_r);
  fillLn(c, a.ln_s);
  fillLin(c, a.o); fillLin(c, a.q); fillLin(c, a.s); fillLin(c, a.v);
  if (has_r) fillLin(c, a.vr);
}
static void fillFour(Cur& c, FourP& f, int din){
  f.b1 = c.nf(); f.b2 = c.nf(); f.freqs = c.nf(); f.lnb = c.nf();
  fillLn(c, f.lno); f.lnw = c.nf(); fillLin(c, f.out);
  f.w1 = c.nf(); f.w2 = c.nf(); f.din = din;
}
static void fillMlp(Cur& c, MlpP& m){ fillLin(c, m.l1); fillLin(c, m.l2); fillLn(c, m.ln); }
static void fillGru(Cur& c, GruP& g){ g.bh = c.nf(); g.bi = c.nf(); g.h0 = c.nf(); g.wh = c.nf(); g.wi = c.nf(); }

static inline dim3 GS(long n, int b){ return dim3((unsigned)((n + b - 1) / b)); }

static void GE(hipStream_t st, const float* A, int lda, const float* B, int ldb,
               const float* bias, float* C, int ldc, int M, int N, int K,
               int act, bool acc){
  dim3 g((N + 63) / 64, (M + 15) / 16), b(128);
  if (!acc) {
    if (act == 0)      gemm_wmma_k<0,false><<<g,b,0,st>>>(A,lda,B,ldb,bias,C,ldc,M,N,K);
    else if (act == 1) gemm_wmma_k<1,false><<<g,b,0,st>>>(A,lda,B,ldb,bias,C,ldc,M,N,K);
    else               gemm_wmma_k<2,false><<<g,b,0,st>>>(A,lda,B,ldb,bias,C,ldc,M,N,K);
  } else {
    if (act == 0)      gemm_wmma_k<0,true ><<<g,b,0,st>>>(A,lda,B,ldb,bias,C,ldc,M,N,K);
    else if (act == 1) gemm_wmma_k<1,true ><<<g,b,0,st>>>(A,lda,B,ldb,bias,C,ldc,M,N,K);
    else               gemm_wmma_k<2,true ><<<g,b,0,st>>>(A,lda,B,ldb,bias,C,ldc,M,N,K);
  }
}

extern "C" void kernel_launch(void* const* d_in, const int* in_sizes, int n_in,
                              void* d_out, int out_size, void* d_ws, size_t ws_size,
                              hipStream_t stream) {
  const float* x_a   = (const float*)d_in[0];   // [A,T,D] == x_t [NT,128]
  const float* x_pl  = (const float*)d_in[1];   // [PL,D]
  const float* rt_in = (const float*)d_in[2];   // [E1,4]
  const float* rp_in = (const float*)d_in[3];   // [E2,3]
  const float* ra_in = (const float*)d_in[4];   // [E3,3]
  const int* e1s = (const int*)d_in[5];  const int* e1d = (const int*)d_in[6];
  const int* e2s = (const int*)d_in[7];  const int* e2d = (const int*)d_in[8];
  const int* e3s = (const int*)d_in[9];  const int* e3d = (const int*)d_in[10];
  const int E1 = in_sizes[5], E2 = in_sizes[7], E3 = in_sizes[9];
  const int Emax = E1 > E2 ? (E1 > E3 ? E1 : E3) : (E2 > E3 ? E2 : E3);

  // ---- parameters (jax tree order: sorted dict keys) ----
  Cur cur{d_in, 11};
  AttnP a2m_p[KL], a2m_r[KL], pl2m_p[KL], pl2m_r[KL], t2m_p[KL], t2m_r[KL], m2m_p, m2m_r;
  GruP gru; MlpP loc_p, loc_r, pi_p, scale_p, scale_r;
  FourP f_a2m, f_pl2m, f_t2m, f_y;
  const float* mode_emb;
  for (int i = 0; i < KL; i++) fillAttn(cur, a2m_p[i], true);
  for (int i = 0; i < KL; i++) fillAttn(cur, a2m_r[i], true);
  fillGru(cur, gru);
  fillMlp(cur, loc_p); fillMlp(cur, loc_r);
  fillAttn(cur, m2m_p, false); fillAttn(cur, m2m_r, false);
  mode_emb = cur.nf();
  fillMlp(cur, pi_p);
  for (int i = 0; i < KL; i++) fillAttn(cur, pl2m_p[i], true);
  for (int i = 0; i < KL; i++) fillAttn(cur, pl2m_r[i], true);
  fillFour(cur, f_a2m, 3); fillFour(cur, f_pl2m, 3); fillFour(cur, f_t2m, 4);
  fillMlp(cur, scale_p); fillMlp(cur, scale_r);
  for (int i = 0; i < KL; i++) fillAttn(cur, t2m_p[i], true);
  for (int i = 0; i < KL; i++) fillAttn(cur, t2m_r[i], true);
  fillFour(cur, f_y, 2);

  // ---- workspace bump allocator ----
  char* wsb = (char*)d_ws; size_t off = 0;
  auto alloc = [&](size_t nfloats)->float* {
    float* p = (float*)(wsb + off);
    off += ((nfloats * sizeof(float) + 255) / 256) * 256;
    return p;
  };
  float* rt   = alloc((size_t)E1 * 128);
  float* rpl  = alloc((size_t)E2 * 128);
  float* ra   = alloc((size_t)E3 * 128);
  float* yemb = alloc((size_t)KM * KFUT * 128);
  float* xplr = alloc((size_t)KK * KPL * 128);
  float* xar  = alloc((size_t)KM * 128);
  float* m    = alloc((size_t)KM * 128);
  float* xsln = alloc((size_t)KNT * 128);
  float* xdln = alloc((size_t)KM * 128);
  float* qb   = alloc((size_t)KM * 128);
  float* sb   = alloc((size_t)KM * 128);
  float* gb   = alloc((size_t)KM * 128);
  float* ob   = alloc((size_t)KM * 128);
  float* aggb = alloc((size_t)KM * 128);
  float* kb   = alloc((size_t)KNT * 128);
  float* vb   = alloc((size_t)KNT * 128);
  float* rln  = alloc((size_t)Emax * 128);
  float* rkb  = alloc((size_t)Emax * 128);
  float* rvb  = alloc((size_t)Emax * 128);
  float* simb = alloc((size_t)Emax * 8);
  float* exb  = alloc((size_t)Emax * 8);
  unsigned* umaxb = (unsigned*)alloc((size_t)KM * 8);
  float* denb = alloc((size_t)KM * 8);
  float* catb = alloc((size_t)KM * 256);
  float* t512 = alloc((size_t)KM * 512);
  float* t1   = alloc((size_t)KM * 128);
  float* t2   = alloc((size_t)KM * 128);
  float* gib  = alloc((size_t)KM * 384);
  float* ghb  = alloc((size_t)KM * 384);
  const int CH = 16384;
  float* featb = alloc((size_t)CH * 129);
  float* hb    = alloc((size_t)CH * 128);
  float* hb2   = alloc((size_t)CH * 128);
  float* sumb  = alloc((size_t)CH * 128);
  float* locsb   = alloc((size_t)KM * 120);
  float* scalesb = alloc((size_t)KM * 120);
  float* tmp120  = alloc((size_t)KM * 120);
  (void)ws_size; (void)n_in; (void)out_size;

  float* out = (float*)d_out;
  float* o_locp   = out;
  float* o_scalep = out + (size_t)KM * 120;
  float* o_locr   = out + (size_t)KM * 240;
  float* o_scaler = out + (size_t)KM * 360;
  float* o_pi     = out + (size_t)KM * 480;

  auto LN = [&](const float* x, const LnP& p, float* y, int rows, int relu){
    ln128_k<<<dim3(rows), dim3(128), 0, stream>>>(x, p.w, p.b, y, relu);
  };
  auto ZERO = [&](float* p, long n){ fill_k<<<GS(n,256),256,0,stream>>>(p, n); };

  // ---- fourier embedding: out[N,128] ----
  auto FOURIER = [&](const FourP& fp, const float* X, int N, float* outp){
    for (int base = 0; base < N; base += CH) {
      int cnt = (N - base) < CH ? (N - base) : CH;
      ZERO(sumb, (long)cnt * 128);
      for (int i = 0; i < fp.din; i++) {
        four_feat_k<<<GS((long)cnt*129,256),256,0,stream>>>(X, fp.din, i, fp.freqs, featb, base, (long)cnt*129);
        GE(stream, featb, 129, fp.w1 + (size_t)i*129*128, 128, fp.b1 + (size_t)i*128,
           hb, 128, cnt, 128, 129, 0, false);
        LnP lw{fp.lnw + (size_t)i*128, fp.lnb + (size_t)i*128};
        LN(hb, lw, hb2, cnt, 1);
        GE(stream, hb2, 128, fp.w2 + (size_t)i*128*128, 128, fp.b2 + (size_t)i*128,
           sumb, 128, cnt, 128, 128, 0, true);
      }
      LN(sumb, fp.lno, hb, cnt, 1);
      GE(stream, hb, 128, fp.out.w, 128, fp.out.b,
         outp + (size_t)base*128, 128, cnt, 128, 128, 0, false);
    }
  };

  // ---- edge attention block, updates m in place ----
  auto EDGE = [&](const AttnP& p, const float* x_src, int Ns, const float* r_emb,
                  const int* src, const int* dst, int E){
    LN(x_src, p.ln_s, xsln, Ns, 0);
    LN(m,     p.ln_d, xdln, KM, 0);
    GE(stream, xdln, 128, p.q.w, 128, p.q.b, qb, 128, KM, 128, 128, 0, false);
    GE(stream, xsln, 128, p.k,   128, nullptr, kb, 128, Ns, 128, 128, 0, false);
    GE(stream, xsln, 128, p.v.w, 128, p.v.b, vb, 128, Ns, 128, 128, 0, false);
    GE(stream, xdln, 128, p.s.w, 128, p.s.b, sb, 128, KM, 128, 128, 0, false);
    LN(r_emb, p.ln_r, rln, E, 0);
    GE(stream, rln, 128, p.kr,   128, nullptr, rkb, 128, E, 128, 128, 0, false);
    GE(stream, rln, 128, p.vr.w, 128, p.vr.b, rvb, 128, E, 128, 128, 0, false);
    ZERO((float*)umaxb, (long)KM * 8);
    ZERO(denb, (long)KM * 8);
    ZERO(aggb, (long)KM * 128);
    long EH = (long)E * 8;
    edge_sim_k    <<<GS(EH,256),256,0,stream>>>(qb, kb, rkb, src, dst, simb, umaxb, EH);
    edge_exp_k    <<<GS(EH,256),256,0,stream>>>(simb, umaxb, dst, exb, denb, EH);
    edge_scatter_k<<<GS(EH,256),256,0,stream>>>(exb, denb, vb, rvb, src, dst, aggb, EH);
    cat2_k<<<GS((long)KM*256,256),256,0,stream>>>(aggb, sb, catb, (long)KM*256);
    GE(stream, catb, 256, p.g.w, 128, p.g.b, gb, 128, KM, 128, 256, 2, false);
    gate_o_k<<<GS((long)KM*128,256),256,0,stream>>>(aggb, gb, sb, ob, (long)KM*128);
    GE(stream, ob, 128, p.o.w, 128, p.o.b, m, 128, KM, 128, 128, 0, true);   // residual
    LN(m, p.ln_f, xdln, KM, 0);
    GE(stream, xdln, 128, p.f1.w, 512, p.f1.b, t512, 512, KM, 512, 128, 1, false);
    GE(stream, t512, 512, p.f2.w, 128, p.f2.b, m, 128, KM, 128, 512, 0, true); // residual
  };

  // ---- dense m2m self attention, updates m in place ----
  auto M2M = [&](const AttnP& p){
    LN(m, p.ln_s, xdln, KM, 0);
    GE(stream, xdln, 128, p.q.w, 128, p.q.b, qb, 128, KM, 128, 128, 0, false);
    GE(stream, xdln, 128, p.k,   128, nullptr, kb, 128, KM, 128, 128, 0, false);
    GE(stream, xdln, 128, p.v.w, 128, p.v.b, vb, 128, KM, 128, 128, 0, false);
    GE(stream, xdln, 128, p.s.w, 128, p.s.b, sb, 128, KM, 128, 128, 0, false);
    m2m_attn_k<<<GS((long)KA*KK*KH,256),256,0,stream>>>(qb, kb, vb, aggb);
    cat2_k<<<GS((long)KM*256,256),256,0,stream>>>(aggb, sb, catb, (long)KM*256);
    GE(stream, catb, 256, p.g.w, 128, p.g.b, gb, 128, KM, 128, 256, 2, false);
    gate_o_k<<<GS((long)KM*128,256),256,0,stream>>>(aggb, gb, sb, ob, (long)KM*128);
    GE(stream, ob, 128, p.o.w, 128, p.o.b, m, 128, KM, 128, 128, 0, true);
    LN(m, p.ln_f, xdln, KM, 0);
    GE(stream, xdln, 128, p.f1.w, 512, p.f1.b, t512, 512, KM, 512, 128, 1, false);
    GE(stream, t512, 512, p.f2.w, 128, p.f2.b, m, 128, KM, 128, 512, 0, true);
  };

  auto MLP = [&](const MlpP& p, float* outp, int ldcOut, int dout){
    GE(stream, m, 128, p.l1.w, 128, p.l1.b, t1, 128, KM, 128, 128, 0, false);
    LN(t1, p.ln, t2, KM, 1);
    GE(stream, t2, 128, p.l2.w, dout, p.l2.b, outp, ldcOut, KM, dout, 128, 0, false);
  };

  // ================= forward =================
  tile_rows_k<<<GS((long)KK*KPL*128,256),256,0,stream>>>(x_pl, xplr, KPL, 128, 0, (long)KK*KPL*128);
  tile_rows_k<<<GS((long)KM*128,256),256,0,stream>>>(x_a, xar, KA, KT*128, (KT-1)*128, (long)KM*128);
  tile_rows_k<<<GS((long)KM*128,256),256,0,stream>>>(mode_emb, m, KK, 128, 0, (long)KM*128);

  FOURIER(f_t2m,  rt_in, E1, rt);
  FOURIER(f_pl2m, rp_in, E2, rpl);
  FOURIER(f_a2m,  ra_in, E3, ra);

  for (int r = 0; r < KREC; r++) {
    for (int i = 0; i < KL; i++) {
      EDGE(t2m_p[i],  x_a,  KNT,     rt,  e1s, e1d, E1);
      EDGE(pl2m_p[i], xplr, KK*KPL,  rpl, e2s, e2d, E2);
      EDGE(a2m_p[i],  xar,  KM,      ra,  e3s, e3d, E3);
    }
    M2M(m2m_p);
    MLP(loc_p,   locsb   + r * 40, 120, 40);
    MLP(scale_p, scalesb + r * 40, 120, 40);
  }
  cumsum_k<<<GS((long)KM*2,256),256,0,stream>>>(locsb,   o_locp,   0, KM);
  cumsum_k<<<GS((long)KM*2,256),256,0,stream>>>(scalesb, o_scalep, 1, KM);

  FOURIER(f_y, o_locp, KM * KFUT, yemb);
  tile_rows_k<<<GS((long)KM*128,256),256,0,stream>>>(gru.h0, m, 1, 0, 0, (long)KM*128);
  for (int t = 0; t < KFUT; t++) {
    GE(stream, yemb + (size_t)t * 128, KFUT * 128, gru.wi, 384, gru.bi, gib, 384, KM, 384, 128, 0, false);
    GE(stream, m, 128, gru.wh, 384, gru.bh, ghb, 384, KM, 384, 128, 0, false);
    gru_gate_k<<<GS((long)KM*128,256),256,0,stream>>>(gib, ghb, m, (long)KM*128);
  }

  for (int i = 0; i < KL; i++) {
    EDGE(t2m_r[i],  x_a,  KNT,    rt,  e1s, e1d, E1);
    EDGE(pl2m_r[i], xplr, KK*KPL, rpl, e2s, e2d, E2);
    EDGE(a2m_r[i],  xar,  KM,     ra,  e3s, e3d, E3);
  }
  M2M(m2m_r);

  MLP(loc_r, tmp120, 120, 120);
  add_k<<<GS((long)KM*120,256),256,0,stream>>>(tmp120, o_locp, o_locr, (long)KM*120);
  MLP(scale_r, tmp120, 120, 120);
  elu11_k<<<GS((long)KM*120,256),256,0,stream>>>(tmp120, o_scaler, (long)KM*120);
  MLP(pi_p, o_pi, 1, 1);
}
